// TransTransformer_13821204758621
// MI455X (gfx1250) — matrix operand order repaired and checked by reading, not compiled, not run on hardware
//
#include <hip/hip_runtime.h>
#include <hip/hip_bf16.h>
#include <math.h>

// ---------------------------------------------------------------------------
// CDNA5 (gfx1250) transformer decoder forward.
// All GEMMs go through v_wmma_f32_16x16x32_bf16 (f32 data packed to bf16 pairs
// with v_perm_b32 while staging tiles into LDS; f32 accumulate).
// ---------------------------------------------------------------------------

typedef __attribute__((ext_vector_type(16))) __bf16 v16bf;
typedef __attribute__((ext_vector_type(2)))  __bf16 v2bf;
typedef __attribute__((ext_vector_type(8)))  float  v8f;

union FragAB { v16bf v; unsigned u[8]; };

__device__ __forceinline__ unsigned short f2bf(float f) {
  union { float f; unsigned u; } x; x.f = f;
  unsigned u = x.u + 0x7FFFu + ((x.u >> 16) & 1u);   // round-to-nearest-even
  return (unsigned short)(u >> 16);
}
// pack two f32 -> two bf16 in one dword
__device__ __forceinline__ unsigned pk2(float a, float b) {
#if __has_builtin(__builtin_amdgcn_cvt_pk_bf16_f32)
  union { v2bf v; unsigned u; } x;
  x.v = __builtin_amdgcn_cvt_pk_bf16_f32(a, b);
  return x.u;
#elif __has_builtin(__builtin_amdgcn_perm)
  // round-half-up then pack high halves with one v_perm_b32:
  // result[15:0] = (a+0x8000)[31:16], result[31:16] = (b+0x8000)[31:16]
  union { float f; unsigned u; } xa, xb;
  xa.f = a; xb.f = b;
  return __builtin_amdgcn_perm(xb.u + 0x8000u, xa.u + 0x8000u, 0x07060302u);
#else
  return (unsigned)f2bf(a) | ((unsigned)f2bf(b) << 16);
#endif
}

#define BM 128
#define BN 128
#define BK 32
#define LSTR 40          // LDS row stride (bf16 units), padded vs 32
#define LSTRW (LSTR / 2) // same stride in dwords

// C[bo,bi] = alpha * A[bo,bi] x B[bo,bi] (+Res) (+ReLU)
// A: MxK row-major (lda). B: KxN row-major (ldb) or, if transB, NxK (ldb).
__global__ __launch_bounds__(256) void wmma_gemm(
    const float* __restrict__ A, const float* __restrict__ B,
    float* __restrict__ C, const float* __restrict__ Res,
    int M, int N, int K, int lda, int ldb, int ldc,
    long long sAo, long long sAi, long long sBo, long long sBi,
    long long sCo, long long sCi, int innerCount,
    float alpha, int transB, int doRelu)
{
  __shared__ unsigned short As[BM * LSTR];
  __shared__ unsigned short Bs[BN * LSTR];   // stored transposed: Bs[n][k]

  const int tid  = threadIdx.x;
  const int bo   = blockIdx.z / innerCount;
  const int bi   = blockIdx.z % innerCount;
  const float* Ab = A + bo * sAo + bi * sAi;
  const float* Bb = B + bo * sBo + bi * sBi;
  float*       Cb = C + bo * sCo + bi * sCi;
  const float* Rb = Res ? (Res + bo * sCo + bi * sCi) : (const float*)0;

  const int m0 = blockIdx.y * BM;
  const int n0 = blockIdx.x * BN;

  const int lane  = tid & 31;
  const int wid   = tid >> 5;      // 8 waves
  const int wr    = wid & 3;       // 4 wave-rows * 32
  const int wc    = wid >> 2;      // 2 wave-cols * 64
  const int lrow  = lane & 15;
  const int lhalf = lane >> 4;

  v8f zero = {};
  v8f acc[2][4];
  #pragma unroll
  for (int mi = 0; mi < 2; ++mi)
    #pragma unroll
    for (int ni = 0; ni < 4; ++ni) acc[mi][ni] = zero;

  for (int k0 = 0; k0 < K; k0 += BK) {
    __syncthreads();
    // ---- stage A tile: 128 rows x 32 cols, f32 -> bf16 (packed) ----
    {
      int row = tid >> 1, seg = (tid & 1) << 4;
      const float* src = Ab + (size_t)(m0 + row) * lda + k0 + seg;
      if (k0 + BK < K) __builtin_prefetch(src + BK, 0, 3);  // next K tile
      unsigned* dst = (unsigned*)&As[row * LSTR + seg];
      #pragma unroll
      for (int i = 0; i < 4; ++i) {
        float4 f = *(const float4*)(src + 4 * i);
        dst[2 * i + 0] = pk2(f.x, f.y);
        dst[2 * i + 1] = pk2(f.z, f.w);
      }
    }
    // ---- stage B tile into Bs[n][k] ----
    if (transB) {
      int n = tid >> 1, seg = (tid & 1) << 4;
      const float* src = Bb + (size_t)(n0 + n) * ldb + k0 + seg;
      if (k0 + BK < K) __builtin_prefetch(src + BK, 0, 3);
      unsigned* dst = (unsigned*)&Bs[n * LSTR + seg];
      #pragma unroll
      for (int i = 0; i < 4; ++i) {
        float4 f = *(const float4*)(src + 4 * i);
        dst[2 * i + 0] = pk2(f.x, f.y);
        dst[2 * i + 1] = pk2(f.z, f.w);
      }
    } else {
      // each thread: 2 adjacent K rows x 8 columns -> 8 packed dword stores
      int kp   = tid >> 4;          // K pair index: k = 2*kp
      int nseg = (tid & 15) << 3;   // 8-column segment
      const float* s0 = Bb + (size_t)(k0 + 2 * kp) * ldb + n0 + nseg;
      const float* s1 = s0 + ldb;
      if (k0 + BK < K) __builtin_prefetch(s0 + (size_t)BK * ldb, 0, 3);
      float4 a0 = *(const float4*)(s0);
      float4 a1 = *(const float4*)(s0 + 4);
      float4 b0 = *(const float4*)(s1);
      float4 b1 = *(const float4*)(s1 + 4);
      unsigned* B32 = (unsigned*)Bs;
      B32[(nseg + 0) * LSTRW + kp] = pk2(a0.x, b0.x);
      B32[(nseg + 1) * LSTRW + kp] = pk2(a0.y, b0.y);
      B32[(nseg + 2) * LSTRW + kp] = pk2(a0.z, b0.z);
      B32[(nseg + 3) * LSTRW + kp] = pk2(a0.w, b0.w);
      B32[(nseg + 4) * LSTRW + kp] = pk2(a1.x, b1.x);
      B32[(nseg + 5) * LSTRW + kp] = pk2(a1.y, b1.y);
      B32[(nseg + 6) * LSTRW + kp] = pk2(a1.z, b1.z);
      B32[(nseg + 7) * LSTRW + kp] = pk2(a1.w, b1.w);
    }
    __syncthreads();

    // ---- load fragments (documented 16-bit A/B VGPR layouts) ----
    FragAB afr[2], bfr[4];
    #pragma unroll
    for (int mi = 0; mi < 2; ++mi) {
      int mrow = wr * 32 + mi * 16 + lrow;
      #pragma unroll
      for (int v = 0; v < 8; ++v) {
        int kl = (v < 4 ? 2 * v : 16 + 2 * (v - 4)) + 8 * lhalf;
        afr[mi].u[v] = *(const unsigned*)&As[mrow * LSTR + kl];
      }
    }
    #pragma unroll
    for (int ni = 0; ni < 4; ++ni) {
      int nrow = wc * 64 + ni * 16 + lrow;
      #pragma unroll
      for (int v = 0; v < 8; ++v) {
        int kl = (v < 4 ? 2 * v : 16 + 2 * (v - 4)) + 8 * lhalf;
        bfr[ni].u[v] = *(const unsigned*)&Bs[nrow * LSTR + kl];
      }
    }
    #pragma unroll
    for (int mi = 0; mi < 2; ++mi)
      #pragma unroll
      for (int ni = 0; ni < 4; ++ni)
        acc[mi][ni] = __builtin_amdgcn_wmma_f32_16x16x32_bf16(
            false, afr[mi].v, false, bfr[ni].v, (short)0, acc[mi][ni],
            false, false);
  }

  // ---- epilogue: scale, relu, residual, store f32 ----
  #pragma unroll
  for (int mi = 0; mi < 2; ++mi) {
    #pragma unroll
    for (int ni = 0; ni < 4; ++ni) {
      #pragma unroll
      for (int r = 0; r < 8; ++r) {
        int grow = m0 + wr * 32 + mi * 16 + r + 8 * lhalf;
        int gcol = n0 + wc * 64 + ni * 16 + lrow;
        float v = acc[mi][ni][r] * alpha;
        if (doRelu) v = fmaxf(v, 0.f);
        size_t idx = (size_t)grow * ldc + gcol;
        if (Rb) v += Rb[idx];
        Cb[idx] = v;
      }
    }
  }
}

// ---------------------------------------------------------------------------
// x[b,0,:] = sos ; x[b,t,:768] = tok_emb[gold[b,t-1]] ; x[b,t,768+j] = (j==t-1)
__global__ void embed_kernel(const int* __restrict__ toks,
                             const float* __restrict__ temb,
                             const float* __restrict__ sos,
                             float* __restrict__ x, int T, int D, int EMB)
{
  int row = blockIdx.x;
  int b = row / T, t = row % T;
  float* xr = x + (size_t)row * D;
  if (t == 0) {
    for (int d = threadIdx.x; d < D; d += blockDim.x) xr[d] = sos[d];
  } else {
    int tok = toks[(size_t)b * T + (t - 1)];
    const float* er = temb + (size_t)tok * EMB;
    for (int d = threadIdx.x; d < D; d += blockDim.x)
      xr[d] = (d < EMB) ? er[d] : ((d - EMB) == (t - 1) ? 1.f : 0.f);
  }
}

// LayerNorm over D=1024, one block (256 threads) per row.
__global__ __launch_bounds__(256) void ln1024(const float* __restrict__ x,
                                              float* __restrict__ y)
{
  __shared__ float red[256];
  int row = blockIdx.x, tid = threadIdx.x;
  const float* xr = x + (size_t)row * 1024;
  float v[4]; float s = 0.f;
  #pragma unroll
  for (int i = 0; i < 4; ++i) { v[i] = xr[tid + 256 * i]; s += v[i]; }
  red[tid] = s; __syncthreads();
  for (int st = 128; st > 0; st >>= 1) {
    if (tid < st) red[tid] += red[tid + st];
    __syncthreads();
  }
  float mu = red[0] * (1.f / 1024.f); __syncthreads();
  float q = 0.f;
  #pragma unroll
  for (int i = 0; i < 4; ++i) { float d = v[i] - mu; q += d * d; }
  red[tid] = q; __syncthreads();
  for (int st = 128; st > 0; st >>= 1) {
    if (tid < st) red[tid] += red[tid + st];
    __syncthreads();
  }
  float rinv = rsqrtf(red[0] * (1.f / 1024.f) + 1e-6f);
  float* yr = y + (size_t)row * 1024;
  #pragma unroll
  for (int i = 0; i < 4; ++i) yr[tid + 256 * i] = (v[i] - mu) * rinv;
}

// Softmax over rows of length 256 (in place); optional causal mask.
__global__ __launch_bounds__(256) void softmax256(float* __restrict__ s, int causal)
{
  __shared__ float red[256];
  int row = blockIdx.x, tid = threadIdx.x;
  int q = row & 255;
  int masked = (causal && tid > q);
  float v = s[(size_t)row * 256 + tid];
  if (masked) v = -__builtin_inff();
  red[tid] = v; __syncthreads();
  for (int st = 128; st > 0; st >>= 1) {
    if (tid < st) red[tid] = fmaxf(red[tid], red[tid + st]);
    __syncthreads();
  }
  float m = red[0]; __syncthreads();
  float e = masked ? 0.f : __expf(v - m);
  red[tid] = e; __syncthreads();
  for (int st = 128; st > 0; st >>= 1) {
    if (tid < st) red[tid] += red[tid + st];
    __syncthreads();
  }
  s[(size_t)row * 256 + tid] = e / red[0];
}

// log_softmax over vocab (in place), one block per row.
__global__ __launch_bounds__(256) void logsoftmax_v(float* __restrict__ out, int V)
{
  __shared__ float red[256];
  int row = blockIdx.x, tid = threadIdx.x;
  float* r = out + (size_t)row * V;
  float m = -__builtin_inff();
  for (int j = tid; j < V; j += 256) m = fmaxf(m, r[j]);
  red[tid] = m; __syncthreads();
  for (int st = 128; st > 0; st >>= 1) {
    if (tid < st) red[tid] = fmaxf(red[tid], red[tid + st]);
    __syncthreads();
  }
  m = red[0]; __syncthreads();
  float s = 0.f;
  for (int j = tid; j < V; j += 256) s += __expf(r[j] - m);
  red[tid] = s; __syncthreads();
  for (int st = 128; st > 0; st >>= 1) {
    if (tid < st) red[tid] += red[tid + st];
    __syncthreads();
  }
  float lg = m + __logf(red[0]);
  for (int j = tid; j < V; j += 256) r[j] = r[j] - lg;
}

// ---------------------------------------------------------------------------
extern "C" void kernel_launch(void* const* d_in, const int* in_sizes, int n_in,
                              void* d_out, int out_size, void* d_ws, size_t ws_size,
                              hipStream_t stream)
{
  (void)in_sizes; (void)n_in; (void)out_size; (void)ws_size;

  const float* memory = (const float*)d_in[0];   // (8,256,1024)
  const int*   gold   = (const int*)  d_in[1];   // (8,256)
  const float* attn_w = (const float*)d_in[2];   // (6,2,4,1024,1024)
  const float* ffn_w1 = (const float*)d_in[3];   // (6,1024,4096)
  const float* ffn_w2 = (const float*)d_in[4];   // (6,4096,1024)
  const float* tokemb = (const float*)d_in[5];   // (32000,768)
  const float* sos    = (const float*)d_in[6];   // (1024,)
  const float* gen_w  = (const float*)d_in[7];   // (1024,32000)

  const int Bsz = 8, ML = 256, D = 1024, NL = 6, H = 8, DK = 128, V = 32000;
  const int EMB = D - ML;            // 768
  const int Mtok = Bsz * ML;         // 2048
  const size_t DD = (size_t)D * D;

  // workspace carve-out (f32): ~96 MB total
  float* xf  = (float*)d_ws;                                  // 2048x1024
  float* lnb = xf  + (size_t)Mtok * D;                        // 2048x1024
  float* qb  = lnb + (size_t)Mtok * D;                        // 2048x1024
  float* kb  = qb  + (size_t)Mtok * D;                        // 2048x1024
  float* vb  = kb  + (size_t)Mtok * D;                        // 2048x1024
  float* ob  = vb  + (size_t)Mtok * D;                        // 2048x1024
  float* sc  = ob  + (size_t)Mtok * D;                        // 64x256x256
  float* ff  = sc  + (size_t)Bsz * H * ML * ML;               // 2048x4096

  auto gemm = [&](const float* A, const float* Bm, float* C, const float* Res,
                  int M, int N, int K, int lda, int ldb, int ldc,
                  long long sAo, long long sAi, long long sBo, long long sBi,
                  long long sCo, long long sCi, int bo, int bi,
                  float alpha, int transB, int relu) {
    dim3 grid(N / BN, M / BM, bo * bi);
    wmma_gemm<<<grid, 256, 0, stream>>>(A, Bm, C, Res, M, N, K, lda, ldb, ldc,
                                        sAo, sAi, sBo, sBi, sCo, sCi, bi,
                                        alpha, transB, relu);
  };

  // one multi-head attention block: q from lnb, k/v from kv_in; x += attn
  auto attention = [&](const float* kv_in, const float* W, int causal) {
    const float* Wq = W;             const float* Wk = W + DD;
    const float* Wv = W + 2 * DD;    const float* Wo = W + 3 * DD;
    gemm(lnb,   Wq, qb, 0, Mtok, D, D, D, D, D, 0,0,0,0,0,0, 1,1, 1.f, 0, 0);
    gemm(kv_in, Wk, kb, 0, Mtok, D, D, D, D, D, 0,0,0,0,0,0, 1,1, 1.f, 0, 0);
    gemm(kv_in, Wv, vb, 0, Mtok, D, D, D, D, D, 0,0,0,0,0,0, 1,1, 1.f, 0, 0);
    // scores[b,h] = (Q_bh (256x128)) x (K_bh)^T / sqrt(dk)   -> sc (64,256,256)
    gemm(qb, kb, sc, 0, ML, ML, DK, D, D, ML,
         (long long)ML * D, DK, (long long)ML * D, DK,
         (long long)H * ML * ML, (long long)ML * ML, Bsz, H,
         0.08838834764831845f, /*transB=*/1, 0);
    softmax256<<<Bsz * H * ML, 256, 0, stream>>>(sc, causal);
    // O_bh (256x128) = P_bh (256x256) x V_bh (256x128)
    gemm(sc, vb, ob, 0, ML, DK, ML, ML, D, D,
         (long long)H * ML * ML, (long long)ML * ML,
         (long long)ML * D, DK, (long long)ML * D, DK, Bsz, H,
         1.f, 0, 0);
    // x = x + O @ Wo
    gemm(ob, Wo, xf, xf, Mtok, D, D, D, D, D, 0,0,0,0,0,0, 1,1, 1.f, 0, 0);
  };

  // ---- embedding ----
  embed_kernel<<<Mtok, 256, 0, stream>>>(gold, tokemb, sos, xf, ML, D, EMB);

  // ---- decoder layers ----
  for (int l = 0; l < NL; ++l) {
    const float* Wself  = attn_w + ((size_t)l * 2 + 0) * 4 * DD;
    const float* Wcross = attn_w + ((size_t)l * 2 + 1) * 4 * DD;

    ln1024<<<Mtok, 256, 0, stream>>>(xf, lnb);
    attention(lnb, Wself, /*causal=*/1);

    ln1024<<<Mtok, 256, 0, stream>>>(xf, lnb);
    attention(memory, Wcross, /*causal=*/0);

    ln1024<<<Mtok, 256, 0, stream>>>(xf, lnb);
    gemm(lnb, ffn_w1 + (size_t)l * D * 4 * D, ff, 0,
         Mtok, 4 * D, D, D, 4 * D, 4 * D, 0,0,0,0,0,0, 1,1, 1.f, 0, /*relu=*/1);
    gemm(ff, ffn_w2 + (size_t)l * 4 * D * D, xf, xf,
         Mtok, D, 4 * D, 4 * D, D, D, 0,0,0,0,0,0, 1,1, 1.f, 0, 0);
  }

  // ---- generator + log_softmax ----
  ln1024<<<Mtok, 256, 0, stream>>>(xf, lnb);
  gemm(lnb, gen_w, (float*)d_out, 0, Mtok, V, D, D, V, V,
       0,0,0,0,0,0, 1,1, 1.f, 0, 0);
  logsoftmax_v<<<Mtok, 256, 0, stream>>>((float*)d_out, V);
}
